// Transformer_77309411684
// MI455X (gfx1250) — compile-verified, matrix operand order checked by reading, not run on hardware
//
#include <hip/hip_runtime.h>

typedef __attribute__((ext_vector_type(16))) _Float16 v16h;
typedef __attribute__((ext_vector_type(8)))  float    v8f;

#define TM 64
#define TN 128
#define TK 32
#define LDSK 40   // padded LDS row stride in halves: 80B (16B aligned), 20-dword stride for bank spread

enum {
  F_RES = 1, F_RELU = 2, F_WF32 = 4, F_WF16 = 8, F_BIAS = 16,
  F_CTRI = 32,   // skip output blocks strictly above the causal diagonal
  F_CKLIM = 64   // limit K to bm+TM (A columns beyond the diagonal are exactly zero)
};

// ---------------- TDM staging path (guarded: device pass + builtin present) ----------------
#if defined(__gfx1250__) && __has_builtin(__builtin_amdgcn_tensor_load_to_lds)
#define USE_TDM 1
typedef __attribute__((ext_vector_type(4))) unsigned tdm_u4;
typedef __attribute__((ext_vector_type(8))) int      tdm_i8;
typedef __attribute__((ext_vector_type(4))) int      tdm_i4;

// DMA one [rows x 32-half] f16 tile (row stride = ld elements) into LDS at lds_byte_off,
// padding 4 dwords after every 16 dwords -> LDS row stride of 40 halves (LDSK).
__device__ inline void tdm_load_f16_tile(const _Float16* gsrc, unsigned lds_byte_off,
                                         unsigned rows, unsigned ld) {
  unsigned long long ga = (unsigned long long)gsrc;
  tdm_u4 g0;
  g0[0] = 1u;                                             // count=1 (valid), user mode
  g0[1] = lds_byte_off;                                   // lds_addr
  g0[2] = (unsigned)(ga & 0xFFFFFFFFull);                 // global_addr[31:0]
  g0[3] = (unsigned)((ga >> 32) & 0x01FFFFFFull) | (2u << 30);  // addr[56:32], type=2
  tdm_i8 g1;
  // data_size=2B (code 1), pad_enable, pad_interval=16 dwords (code 3), pad_amount=4 dwords (code 3)
  g1[0] = (1 << 16) | (1 << 20) | (3 << 22) | (3 << 25);
  const unsigned td0 = ld;              // tensor_dim0 (elements) -> no OOB within a row
  const unsigned td1 = 0x7FFFFFFFu;     // tensor_dim1: huge, no OOB clipping
  g1[1] = (int)((td0 & 0xFFFFu) << 16);                        // atomic_addr=0 | td0[15:0]
  g1[2] = (int)((td0 >> 16) | ((td1 & 0xFFFFu) << 16));        // td0[31:16] | td1[15:0]
  g1[3] = (int)((td1 >> 16) | ((unsigned)TK << 16));           // td1[31:16] | tile_dim0=32
  g1[4] = (int)rows;                                           // tile_dim1 | tile_dim2=0
  g1[5] = (int)ld;                                             // tensor_dim0_stride[31:0]
  g1[6] = 0;                                                   // stride[47:32] | dim1_stride lo
  g1[7] = 0;
  tdm_i4 z4 = {0, 0, 0, 0};
  tdm_i8 z8 = {0, 0, 0, 0, 0, 0, 0, 0};
  // 6-arg form (clang-23 / therock headers): (g0, g1, g2, g3, extra, cpol)
  __builtin_amdgcn_tensor_load_to_lds(g0, g1, z4, z4, z8, 0);
}
#else
#define USE_TDM 0
#endif

// Load 8 elements (f16 directly, or f32 converted to f16) into dst[0..7].
template <typename TB>
__device__ inline void load8_f16(const TB* __restrict__ p, _Float16* dst) {
  if constexpr (sizeof(TB) == 2) {
    union { uint4 u; _Float16 h[8]; } t;
    t.u = *(const uint4*)p;
#pragma unroll
    for (int e = 0; e < 8; ++e) dst[e] = t.h[e];
  } else {
    const float4* q = (const float4*)p;
    float4 a = q[0], b = q[1];
    dst[0] = (_Float16)a.x; dst[1] = (_Float16)a.y;
    dst[2] = (_Float16)a.z; dst[3] = (_Float16)a.w;
    dst[4] = (_Float16)b.x; dst[5] = (_Float16)b.y;
    dst[6] = (_Float16)b.z; dst[7] = (_Float16)b.w;
  }
}

// C[M,N](f32 acc) = A[M,K](f16) * B (f16 or f32; NN: [K,N], NT: [N,K])
// Epilogue: optional bias (per-N), relu, residual add from Cf, write f32 and/or f16.
template <bool TRANS_B, typename TB>
__global__ __launch_bounds__(256) void gemm_f16_wmma(
    const _Float16* __restrict__ A, const TB* __restrict__ B,
    float* __restrict__ Cf, _Float16* __restrict__ Ch,
    const float* __restrict__ bias,
    int M, int N, int K, int lda, int ldb,
    long long sA, long long sB, long long sC, int flags) {
  __shared__ _Float16 As[TM][LDSK];
  __shared__ _Float16 Bs[TN][LDSK];

  const int bm = blockIdx.y * TM;
  const int bn = blockIdx.x * TN;

  // Causal: output blocks strictly above the diagonal are never read by softmax.
  if ((flags & F_CTRI) && bn >= bm + TM) return;
  // Causal: A columns beyond the diagonal are exactly zero -> trim the K loop.
  const int Klim = (flags & F_CKLIM) ? ((bm + TM < K) ? (bm + TM) : K) : K;

  const int z = blockIdx.z;
  A += (long long)z * sA;
  B += (long long)z * sB;
  const long long coff = (long long)z * sC;

  const int tid  = threadIdx.x;
  const int lane = tid & 31;
  const int wid  = tid >> 5;
  const int wm   = wid >> 2;   // 0..1 (M)
  const int wn   = wid & 3;    // 0..3 (N)
  const int hf   = lane >> 4;  // 0..1
  const int idx  = lane & 15;  // 0..15

  v8f acc[2][2] = {};

  // Per-thread staging coordinates.
  const int ar  = tid >> 2;        // 0..63   A-tile row
  const int ac  = (tid & 3) * 8;   // 0..24   A-tile k-chunk
  const int br  = tid >> 1;        // 0..127  NT: B rows are N
  const int bc  = (tid & 1) * 16;  // 0/16
  const int bk  = tid >> 3;        // 0..31   NN: B rows are K
  const int bn0 = (tid & 7) * 16;  // 0..112

#if USE_TDM
  const unsigned ldsA = (unsigned)(unsigned long long)&As[0][0];
  const unsigned ldsB = (unsigned)(unsigned long long)&Bs[0][0];
#endif

  for (int k0 = 0; k0 < Klim; k0 += TK) {
    // ---- stage A tile (f16, row-major) ----
#if USE_TDM
    if (wid == 0)
      tdm_load_f16_tile(A + (size_t)bm * lda + k0, ldsA, TM, (unsigned)lda);
#else
    {
      _Float16 t[8];
      load8_f16<_Float16>(A + (size_t)(bm + ar) * lda + k0 + ac, t);
#pragma unroll
      for (int e = 0; e < 8; ++e) As[ar][ac + e] = t[e];
      if (k0 + TK < Klim)
        __builtin_prefetch(A + (size_t)(bm + ar) * lda + k0 + TK + ac, 0, 0);
    }
#endif
    // ---- stage B tile as Bs[n][k] ----
    if constexpr (TRANS_B && sizeof(TB) == 2) {
#if USE_TDM
      if (wid == 1)
        tdm_load_f16_tile((const _Float16*)B + (size_t)bn * ldb + k0, ldsB, TN,
                          (unsigned)ldb);
#else
      _Float16 t[16];
      const TB* p = B + (size_t)(bn + br) * ldb + k0 + bc;
      load8_f16<TB>(p, t);
      load8_f16<TB>(p + 8, t + 8);
#pragma unroll
      for (int e = 0; e < 16; ++e) Bs[br][bc + e] = t[e];
#endif
    } else if constexpr (TRANS_B) {
      _Float16 t[16];
      const TB* p = B + (size_t)(bn + br) * ldb + k0 + bc;
      load8_f16<TB>(p, t);
      load8_f16<TB>(p + 8, t + 8);
#pragma unroll
      for (int e = 0; e < 16; ++e) Bs[br][bc + e] = t[e];
    } else {
      _Float16 t[16];
      const TB* p = B + (size_t)(k0 + bk) * ldb + bn + bn0;
      load8_f16<TB>(p, t);
      load8_f16<TB>(p + 8, t + 8);
#pragma unroll
      for (int e = 0; e < 16; ++e) Bs[bn0 + e][bk] = t[e];
    }
#if USE_TDM
    __builtin_amdgcn_s_wait_tensorcnt(0);  // NOP for waves with no outstanding TDM
#endif
    __syncthreads();

    // ---- build fragments per ISA 16-bit layout: VGPR p <-> K pair {2p,2p+1} (+16 for p>=4), +8*half ----
    v16h afrag[2], bfrag[2];
#pragma unroll
    for (int i = 0; i < 2; ++i) {
      unsigned* ap = (unsigned*)&afrag[i];
      const int row = wm * 32 + i * 16 + idx;
#pragma unroll
      for (int p = 0; p < 8; ++p) {
        const int kk = ((p < 4) ? (p * 2) : (16 + (p - 4) * 2)) + hf * 8;
        ap[p] = *(const unsigned*)&As[row][kk];
      }
    }
#pragma unroll
    for (int j = 0; j < 2; ++j) {
      unsigned* bp = (unsigned*)&bfrag[j];
      const int col = wn * 32 + j * 16 + idx;
#pragma unroll
      for (int p = 0; p < 8; ++p) {
        const int kk = ((p < 4) ? (p * 2) : (16 + (p - 4) * 2)) + hf * 8;
        bp[p] = *(const unsigned*)&Bs[col][kk];
      }
    }

#pragma unroll
    for (int i = 0; i < 2; ++i)
#pragma unroll
      for (int j = 0; j < 2; ++j)
        acc[i][j] = __builtin_amdgcn_wmma_f32_16x16x32_f16(
            false, afrag[i], false, bfrag[j], (short)0, acc[i][j], false, false);
    __syncthreads();
  }

  // ---- epilogue: C VGPR r <-> M = r + 8*half, N = idx ----
#pragma unroll
  for (int i = 0; i < 2; ++i) {
#pragma unroll
    for (int j = 0; j < 2; ++j) {
#pragma unroll
      for (int r = 0; r < 8; ++r) {
        const int row = bm + wm * 32 + i * 16 + hf * 8 + r;
        const int col = bn + wn * 32 + j * 16 + idx;
        float v = acc[i][j][r];
        if (flags & F_BIAS) v += bias[col];
        if (flags & F_RELU) v = fmaxf(v, 0.0f);
        const long long off = coff + (long long)row * N + col;
        if (flags & F_RES)  v += Cf[off];
        if (flags & F_WF32) Cf[off] = v;
        if (flags & F_WF16) Ch[off] = (_Float16)v;
      }
    }
  }
}

__global__ __launch_bounds__(256) void embed_kernel(
    const int* __restrict__ tokens, const float* __restrict__ tok_embed,
    const float* __restrict__ pos_embed, float* __restrict__ X,
    _Float16* __restrict__ Xh, int S, int D) {
  const long long gid = blockIdx.x;  // b*S + s
  const int s = (int)(gid % S);
  const long long tok = tokens[gid];
  const float* te = tok_embed + tok * D;
  const float* pe = pos_embed + (long long)s * D;
  float* xp = X + gid * D;
  _Float16* xh = Xh + gid * D;
  for (int d = threadIdx.x; d < D; d += 256) {
    float v = te[d] + pe[d];
    xp[d] = v;
    xh[d] = (_Float16)v;
  }
}

// One block per (b,s) row: causal-masked softmax over scores row, write f16 probs
// (exact zeros above the diagonal, relied upon by the F_CKLIM GEMM).
__global__ __launch_bounds__(256) void softmax_causal(
    const float* __restrict__ scores, _Float16* __restrict__ Ah, int S) {
  __shared__ float red[256];
  const long long row = blockIdx.x;  // b*S + s
  const int s = (int)(row % S);
  const float* sp = scores + row * (long long)S;
  _Float16* op = Ah + row * (long long)S;
  const int tid = threadIdx.x;

  float m = -3.4e38f;
  for (int j = tid; j <= s; j += 256) m = fmaxf(m, sp[j]);
  red[tid] = m;
  __syncthreads();
  for (int o = 128; o > 0; o >>= 1) {
    if (tid < o) red[tid] = fmaxf(red[tid], red[tid + o]);
    __syncthreads();
  }
  m = red[0];
  __syncthreads();

  float sum = 0.0f;
  for (int j = tid; j <= s; j += 256) sum += __expf(sp[j] - m);
  red[tid] = sum;
  __syncthreads();
  for (int o = 128; o > 0; o >>= 1) {
    if (tid < o) red[tid] += red[tid + o];
    __syncthreads();
  }
  const float inv = 1.0f / red[0];

  for (int j = tid; j < S; j += 256)
    op[j] = (j <= s) ? (_Float16)(__expf(sp[j] - m) * inv) : (_Float16)0.0f;
}

extern "C" void kernel_launch(void* const* d_in, const int* in_sizes, int n_in,
                              void* d_out, int out_size, void* d_ws, size_t ws_size,
                              hipStream_t stream) {
  (void)in_sizes; (void)n_in; (void)out_size; (void)ws_size;
  const int Bn = 2, S = 2048, D = 1024, DMLP = 4096, V = 32000, L = 2;
  const int BS = Bn * S;

  const int*   tokens    = (const int*)d_in[0];
  const float* tok_embed = (const float*)d_in[1];
  const float* pos_embed = (const float*)d_in[2];
  const float* W_QK      = (const float*)d_in[3];
  const float* W_OV      = (const float*)d_in[4];
  const float* W_in_w    = (const float*)d_in[5];
  const float* W_in_b    = (const float*)d_in[6];
  const float* W_out_w   = (const float*)d_in[7];
  const float* W_out_b   = (const float*)d_in[8];
  const float* unembed_w = (const float*)d_in[9];
  float* logits = (float*)d_out;

  // Workspace carve-out (256B aligned).
  char* ws = (char*)d_ws;
  size_t off = 0;
  auto carve = [&](size_t bytes) {
    void* p = ws + off;
    off += (bytes + 255) & ~(size_t)255;
    return p;
  };
  float*    X      = (float*)carve((size_t)BS * D * 4);
  float*    scores = (float*)carve((size_t)Bn * S * S * 4);
  _Float16* Xh     = (_Float16*)carve((size_t)BS * D * 2);
  _Float16* Qh     = (_Float16*)carve((size_t)BS * D * 2);
  _Float16* Ah     = (_Float16*)carve((size_t)Bn * S * S * 2);
  _Float16* AXh    = (_Float16*)carve((size_t)BS * D * 2);
  _Float16* hh     = (_Float16*)carve((size_t)BS * DMLP * 2);

  embed_kernel<<<BS, 256, 0, stream>>>(tokens, tok_embed, pos_embed, X, Xh, S, D);

  const long long llSD = (long long)S * D, llSS = (long long)S * S;

  for (int i = 0; i < L; ++i) {
    const float* wqk  = W_QK    + (size_t)i * D * D;
    const float* wov  = W_OV    + (size_t)i * D * D;
    const float* win  = W_in_w  + (size_t)i * DMLP * D;
    const float* winb = W_in_b  + (size_t)i * DMLP;
    const float* wout = W_out_w + (size_t)i * D * DMLP;
    const float* woutb= W_out_b + (size_t)i * D;

    // Q = X @ W_QK          [BS,D] = [BS,D] x NN [D,D]
    gemm_f16_wmma<false, float><<<dim3(D / TN, BS / TM, 1), 256, 0, stream>>>(
        Xh, wqk, nullptr, Qh, nullptr, BS, D, D, D, D, 0, 0, 0, F_WF16);

    // scores = Q @ X^T      per batch: [S,S] = [S,D] x NT [S,D]; skip blocks above diagonal
    gemm_f16_wmma<true, _Float16><<<dim3(S / TN, S / TM, Bn), 256, 0, stream>>>(
        Qh, Xh, scores, nullptr, nullptr, S, S, D, D, D, llSD, llSD, llSS,
        F_WF32 | F_CTRI);

    // causal softmax -> Ah (f16)
    softmax_causal<<<BS, 256, 0, stream>>>(scores, Ah, S);

    // AX = A @ X            per batch: [S,D] = [S,S] x NN [S,D]; K trimmed to diagonal
    gemm_f16_wmma<false, _Float16><<<dim3(D / TN, S / TM, Bn), 256, 0, stream>>>(
        Ah, Xh, nullptr, AXh, nullptr, S, D, S, S, D, llSS, llSD, llSD,
        F_WF16 | F_CKLIM);

    // X += AX @ W_OV        [BS,D] = [BS,D] x NN [D,D], residual into X, refresh Xh
    gemm_f16_wmma<false, float><<<dim3(D / TN, BS / TM, 1), 256, 0, stream>>>(
        AXh, wov, X, Xh, nullptr, BS, D, D, D, D, 0, 0, 0, F_RES | F_WF32 | F_WF16);

    // h = relu(X @ W_in^T + b)  [BS,DMLP] = [BS,D] x NT [DMLP,D]
    gemm_f16_wmma<true, float><<<dim3(DMLP / TN, BS / TM, 1), 256, 0, stream>>>(
        Xh, win, nullptr, hh, winb, BS, DMLP, D, D, D, 0, 0, 0,
        F_BIAS | F_RELU | F_WF16);

    // X += h @ W_out^T + b  [BS,D] = [BS,DMLP] x NT [D,DMLP], residual, refresh Xh
    gemm_f16_wmma<true, float><<<dim3(D / TN, BS / TM, 1), 256, 0, stream>>>(
        hh, wout, X, Xh, woutb, BS, D, DMLP, DMLP, DMLP, 0, 0, 0,
        F_BIAS | F_RES | F_WF32 | F_WF16);
  }

  // logits = X @ unembed^T   [BS,V] = [BS,D] x NT [V,D]
  gemm_f16_wmma<true, float><<<dim3(V / TN, BS / TM, 1), 256, 0, stream>>>(
      Xh, unembed_w, logits, nullptr, nullptr, BS, V, D, D, D, 0, 0, 0, F_WF32);
}
